// GCN_45595372814849
// MI455X (gfx1250) — matrix operand order repaired and checked by reading, not compiled, not run on hardware
//
#include <hip/hip_runtime.h>
#include <math.h>

#define NNODES 50000
#define NEDGES 800000
#define DIM    128
#define BN_EPS 1e-5f

typedef __attribute__((ext_vector_type(16))) _Float16     v16h;
typedef __attribute__((ext_vector_type(8)))  _Float16     v8h;
typedef __attribute__((ext_vector_type(8)))  float        v8f;
typedef __attribute__((ext_vector_type(4)))  unsigned int u32x4;
typedef __attribute__((ext_vector_type(4)))  int          i32x4;
typedef __attribute__((ext_vector_type(8)))  int          i32x8;

#if defined(__has_builtin)
#  if __has_builtin(__builtin_amdgcn_tensor_load_to_lds)
#    define HAS_TDM 1
#  else
#    define HAS_TDM 0
#  endif
#else
#  define HAS_TDM 0
#endif

__device__ __forceinline__ void atomAddF(float* p, float v) {
    __hip_atomic_fetch_add(p, v, __ATOMIC_RELAXED, __HIP_MEMORY_SCOPE_AGENT);
}

// ---------------------------------------------------------------------------
__global__ void fill_f32(float* __restrict__ p, float v, int n) {
    int i = blockIdx.x * blockDim.x + threadIdx.x;
    if (i < n) p[i] = v;
}

__global__ void cvt_f32_f16(const float* __restrict__ X, _Float16* __restrict__ Y, int n) {
    int i = blockIdx.x * blockDim.x + threadIdx.x;
    if (i < n) Y[i] = (_Float16)X[i];
}

// ---------------------------------------------------------------------------
// Degree (self loops via init fill of 1.0) and deg^-1/2
// ---------------------------------------------------------------------------
__global__ void deg_accum(const int* __restrict__ dst, float* __restrict__ deg, int e) {
    int i = blockIdx.x * blockDim.x + threadIdx.x;
    if (i < e) atomAddF(&deg[dst[i]], 1.0f);
}

__global__ void deg_rsqrt(const float* __restrict__ deg, float* __restrict__ dis, int n) {
    int i = blockIdx.x * blockDim.x + threadIdx.x;
    if (i < n) dis[i] = rsqrtf(deg[i]);   // deg >= 1 always (self loop)
}

// ---------------------------------------------------------------------------
// Pack a 128x128 f32 weight (row major, K x N) into the WMMA B-operand
// layout: [kt 0..3][ntile 0..7][lane 0..31][i 0..15] f16, where
//   k = kt*32 + (lane>>4)*16 + i ; n = ntile*16 + (lane & 15)
// Each lane's 16 f16 = 32 contiguous bytes.
// ---------------------------------------------------------------------------
__global__ void pack_w_f16(const float* __restrict__ W, _Float16* __restrict__ Bp) {
    int idx = blockIdx.x * blockDim.x + threadIdx.x;     // 0 .. 16383
    if (idx >= DIM * DIM) return;
    int i    =  idx        & 15;
    int lane = (idx >> 4)  & 31;
    int nt   = (idx >> 9)  & 7;
    int kt   =  idx >> 12;
    int k = kt * 32 + (lane >> 4) * 16 + i;
    int n = nt * 16 + (lane & 15);
    Bp[idx] = (_Float16)W[k * DIM + n];
}

// ---------------------------------------------------------------------------
// C[N,128] = A16[N,128] @ W[128,128] (+bias)(+relu), f16 in, f32 accumulate.
// block = 256 thr = 8 waves; block owns 16 rows; wave w owns 16-col tile.
// B (32 KB packed f16) is staged into LDS once per block by the Tensor Data
// Mover (wave 0 issues; TENSORcnt wait; workgroup barrier); WMMA B-operands
// are 32B contiguous LDS reads; A comes from global as f16 (one b128 clause).
// Compile-time BIAS/RELU/OUT16 -> straight-line epilogue, no per-element
// branches. N = 50000 = 3125*16 exactly -> EXEC all ones at every WMMA.
// ---------------------------------------------------------------------------
template <bool BIAS, bool RELU, bool OUT16>
__global__ __launch_bounds__(256) void gemm128_wmma(
    const _Float16* __restrict__ A16, const _Float16* __restrict__ Bp,
    const float* __restrict__ bias, float* __restrict__ OutF,
    _Float16* __restrict__ OutH)
{
    __shared__ _Float16 Bl[DIM * DIM];            // 32 KB

#if HAS_TDM
    if ((threadIdx.x >> 5) == 0) {                // one TDM issue per block
        unsigned long long ga = (unsigned long long)(size_t)Bp;
        u32x4 g0 = {};
        g0[0] = 1u;                               // count=1, user descriptor
        g0[1] = (unsigned)(size_t)(void*)Bl;      // lds_addr (low 32 = offset)
        g0[2] = (unsigned)(ga & 0xffffffffu);     // global_addr[31:0]
        g0[3] = (unsigned)((ga >> 32) & 0x1ffffffu) | (2u << 30); // [56:32] | type=2
        i32x8 g1 = {};
        g1[0] = (int)(1u << 16);                  // wg_mask=0, data_size=1 (2B)
        g1[1] = (int)(16384u << 16);              // tensor_dim0[15:0] = 16384
        g1[2] = (int)(1u << 16);                  // tensor_dim1 = 1
        g1[3] = (int)(16384u << 16);              // tile_dim0 = 16384 (1-D tile)
        g1[5] = 16384;                            // tensor_dim0_stride
        i32x4 z4 = {};
#  if __clang_major__ >= 23
        i32x8 z8 = {};
        __builtin_amdgcn_tensor_load_to_lds(g0, g1, z4, z4, z8, 0);
#  else
        __builtin_amdgcn_tensor_load_to_lds(g0, g1, z4, z4, 0);
#  endif
        __builtin_amdgcn_s_wait_tensorcnt(0);
    }
    __syncthreads();
#else
    for (int t = threadIdx.x; t < (DIM * DIM) / 8; t += 256)
        ((v8h*)Bl)[t] = ((const v8h*)Bp)[t];
    __syncthreads();
#endif

    const int lane = threadIdx.x & 31;
    const int wv   = threadIdx.x >> 5;            // n-tile 0..7
    const int lid  = lane & 15;
    const int half = lane >> 4;
    const int m    = blockIdx.x * 16 + lid;       // this lane's A row
    const _Float16* __restrict__ hrow = A16 + (size_t)m * DIM;

    __builtin_prefetch(hrow, 0, 0);               // global_prefetch_b8

    v8f acc = {};
    #pragma unroll
    for (int kt = 0; kt < 4; ++kt) {
        // 16-bit A 16x32 layout (matches packed B): half=0 -> K {0-15}; half=1 -> K {16-31}
        const int kb = kt * 32 + half * 16;
        v8h a0 = *(const v8h*)(hrow + kb);
        v8h a1 = *(const v8h*)(hrow + kb + 8);
        v16h a;
        #pragma unroll
        for (int i = 0; i < 8; ++i) { a[i] = a0[i]; a[8 + i] = a1[i]; }
        v16h b = *(const v16h*)(Bl + (((kt * 8 + wv) * 32 + lane) << 4));
        acc = __builtin_amdgcn_wmma_f32_16x16x32_f16(
            false, a, false, b, (short)0, acc, false, false);
    }

    // C/D layout: VGPR j -> row half*8 + j, column = lid within the n-tile
    const int n  = wv * 16 + lid;
    float bv = 0.0f;
    if (BIAS) bv = bias[n];
    #pragma unroll
    for (int j = 0; j < 8; ++j) {
        size_t row = (size_t)(blockIdx.x * 16 + half * 8 + j);
        float v = acc[j];
        if (BIAS) v += bv;
        if (RELU) v = fmaxf(v, 0.0f);
        if (OUT16) OutH[row * DIM + n] = (_Float16)v;
        else       OutF[row * DIM + n] = v;
    }
}

// ---------------------------------------------------------------------------
// Agg init: bias broadcast + self-loop contribution Hw[r]*dis[r]^2
// ---------------------------------------------------------------------------
__global__ void agg_init(const float* __restrict__ Hw, const float* __restrict__ dis,
                         const float* __restrict__ bias, float* __restrict__ Agg,
                         int total)
{
    int idx = blockIdx.x * blockDim.x + threadIdx.x;
    if (idx >= total) return;
    int r = idx >> 7, c = idx & (DIM - 1);
    float d = dis[r];
    Agg[idx] = bias[c] + Hw[idx] * d * d;
}

// ---------------------------------------------------------------------------
// Edge scatter: one wave per edge; lane handles 4 features (float4).
// ---------------------------------------------------------------------------
__global__ __launch_bounds__(256) void scatter_edges(
    const float* __restrict__ Hw, const int* __restrict__ src,
    const int* __restrict__ dst, const float* __restrict__ dis,
    float* __restrict__ Agg, int e)
{
    int edge = blockIdx.x * 8 + (threadIdx.x >> 5);
    if (edge >= e) return;
    int lane = threadIdx.x & 31;
    int s = src[edge], d = dst[edge];
    float nrm = dis[s] * dis[d];
    float4 v = *(const float4*)(Hw + (size_t)s * DIM + lane * 4);
    float* out = Agg + (size_t)d * DIM + lane * 4;
    atomAddF(out + 0, v.x * nrm);
    atomAddF(out + 1, v.y * nrm);
    atomAddF(out + 2, v.z * nrm);
    atomAddF(out + 3, v.w * nrm);
}

// ---------------------------------------------------------------------------
// BatchNorm stats: coalesced partial column sums + agent-scope atomics.
// ---------------------------------------------------------------------------
__global__ __launch_bounds__(128) void bn_partial(
    const float* __restrict__ X, float* __restrict__ sum, float* __restrict__ sq,
    int nrows, int preRelu)
{
    int c  = threadIdx.x;
    int r0 = blockIdx.x * 256;
    int r1 = r0 + 256 < nrows ? r0 + 256 : nrows;
    float s = 0.0f, q = 0.0f;
    for (int r = r0; r < r1; ++r) {
        float v = X[(size_t)r * DIM + c];
        if (preRelu) v = fmaxf(v, 0.0f);
        s += v; q += v * v;
    }
    atomAddF(&sum[c], s);
    atomAddF(&sq[c], q);
}

__global__ __launch_bounds__(128) void bn_finalize(
    const float* __restrict__ sum, const float* __restrict__ sq,
    float* __restrict__ mean, float* __restrict__ invstd, int nrows)
{
    int c = threadIdx.x;
    float m   = sum[c] / (float)nrows;
    float var = sq[c] / (float)nrows - m * m;
    mean[c]   = m;
    invstd[c] = rsqrtf(var + BN_EPS);
}

// bn_apply writes the f16 activation the next GEMM consumes directly.
__global__ void bn_apply(
    const float* __restrict__ X, const float* __restrict__ mean,
    const float* __restrict__ invstd, const float* __restrict__ gamma,
    const float* __restrict__ beta, _Float16* __restrict__ Y,
    int total, int preRelu, int postRelu)
{
    int idx = blockIdx.x * blockDim.x + threadIdx.x;
    if (idx >= total) return;
    int c = idx & (DIM - 1);
    float v = X[idx];
    if (preRelu) v = fmaxf(v, 0.0f);
    v = (v - mean[c]) * invstd[c] * gamma[c] + beta[c];
    if (postRelu) v = fmaxf(v, 0.0f);
    Y[idx] = (_Float16)v;
}

// ---------------------------------------------------------------------------
// Head: out[i] = sigmoid(h[i,:] . w + b). Wave per node, shuffle reduction.
// ---------------------------------------------------------------------------
__global__ __launch_bounds__(256) void head_k(
    const float* __restrict__ H, const float* __restrict__ w,
    const float* __restrict__ ob, float* __restrict__ out, int n)
{
    int node = blockIdx.x * 8 + (threadIdx.x >> 5);
    if (node >= n) return;
    int lane = threadIdx.x & 31;
    float4 hv = *(const float4*)(H + (size_t)node * DIM + lane * 4);
    float4 wv = *(const float4*)(w + lane * 4);
    float s = hv.x * wv.x + hv.y * wv.y + hv.z * wv.z + hv.w * wv.w;
    #pragma unroll
    for (int off = 16; off > 0; off >>= 1) s += __shfl_xor(s, off, 32);
    if (lane == 0) out[node] = 1.0f / (1.0f + expf(-(s + ob[0])));
}

// ---------------------------------------------------------------------------
extern "C" void kernel_launch(void* const* d_in, const int* in_sizes, int n_in,
                              void* d_out, int out_size, void* d_ws, size_t ws_size,
                              hipStream_t stream)
{
    const float* x      = (const float*)d_in[0];               // [N,128]
    const int*   ei     = (const int*)d_in[1];                 // [2,E]
    const float* conv_w = (const float*)d_in[2];               // [3,128,128]
    const float* conv_b = (const float*)d_in[3];               // [3,128]
    const float* bn_g   = (const float*)d_in[4];               // [2,128]
    const float* bn_b   = (const float*)d_in[5];               // [2,128]
    const float* mlp_w  = (const float*)d_in[6];               // [2,128,128]
    const float* mlp_b  = (const float*)d_in[7];               // [2,128]
    const float* out_w  = (const float*)d_in[8];               // [128]
    const float* out_b  = (const float*)d_in[9];               // [1]
    float* out = (float*)d_out;

    const int* srcI = ei;
    const int* dstI = ei + NEDGES;

    const size_t ND = (size_t)NNODES * DIM;
    float* ws   = (float*)d_ws;
    float* hW   = ws;                  // N*D f32 (gemm output / messages)
    float* agg  = hW + ND;             // N*D f32 (aggregation)
    float* deg  = agg + ND;            // N
    float* dis  = deg + NNODES;        // N
    float* bsum = dis + NNODES;        // D
    float* bsq  = bsum + DIM;          // D
    float* bmn  = bsq + DIM;           // D
    float* binv = bmn + DIM;           // D
    _Float16* act16 = (_Float16*)(((uintptr_t)(binv + DIM) + 255) & ~(uintptr_t)255);
    _Float16* h16b  = act16 + ND;      // N*D f16 (MLP intermediate)
    _Float16* Bp    = h16b + ND;       // 5 * 16384 f16 packed weights
    const int WPACK = DIM * DIM;

    const int TPB   = 256;
    const int totND = (int)ND;
    const int gND   = (totND + TPB - 1) / TPB;          // 25000
    const int gGemm = NNODES / 16;                      // 3125 (exact)
    const int gEdge = (NEDGES + 7) / 8;                 // 100000
    const int gBN   = (NNODES + 255) / 256;             // 196
    const int gNode = (NNODES + 7) / 8;                 // 6250

    // ---- degree / normalization -------------------------------------------
    fill_f32<<<(NNODES + TPB - 1) / TPB, TPB, 0, stream>>>(deg, 1.0f, NNODES);
    deg_accum<<<(NEDGES + TPB - 1) / TPB, TPB, 0, stream>>>(dstI, deg, NEDGES);
    deg_rsqrt<<<(NNODES + TPB - 1) / TPB, TPB, 0, stream>>>(deg, dis, NNODES);

    // ---- pack the 5 weight matrices to f16 WMMA B layout ------------------
    for (int l = 0; l < 3; ++l)
        pack_w_f16<<<WPACK / TPB, TPB, 0, stream>>>(conv_w + (size_t)l * WPACK,
                                                    Bp + (size_t)l * WPACK);
    for (int l = 0; l < 2; ++l)
        pack_w_f16<<<WPACK / TPB, TPB, 0, stream>>>(mlp_w + (size_t)l * WPACK,
                                                    Bp + (size_t)(3 + l) * WPACK);

    // ---- x -> f16 activation ---------------------------------------------
    cvt_f32_f16<<<gND, TPB, 0, stream>>>(x, act16, totND);

    // ---- GCN layers -------------------------------------------------------
    for (int l = 0; l < 3; ++l) {
        gemm128_wmma<false, false, false><<<gGemm, TPB, 0, stream>>>(
            act16, Bp + (size_t)l * WPACK, nullptr, hW, nullptr);
        agg_init<<<gND, TPB, 0, stream>>>(hW, dis, conv_b + l * DIM, agg, totND);
        scatter_edges<<<gEdge, TPB, 0, stream>>>(hW, srcI, dstI, dis, agg, NEDGES);

        // layer 2: relu BEFORE bn, reusing bn index 1 (reference quirk)
        const int bnIdx    = (l < 2) ? l : 1;
        const int preRelu  = (l == 2) ? 1 : 0;
        const int postRelu = (l == 2) ? 0 : 1;
        fill_f32<<<1, TPB, 0, stream>>>(bsum, 0.0f, 2 * DIM);  // bsum+bsq contiguous
        bn_partial<<<gBN, 128, 0, stream>>>(agg, bsum, bsq, NNODES, preRelu);
        bn_finalize<<<1, 128, 0, stream>>>(bsum, bsq, bmn, binv, NNODES);
        bn_apply<<<gND, TPB, 0, stream>>>(agg, bmn, binv, bn_g + bnIdx * DIM,
                                          bn_b + bnIdx * DIM, act16, totND,
                                          preRelu, postRelu);
    }

    // ---- MLP head: relu(h@W + b) x2 (f16 chain), sigmoid(h @ out_w + b) ---
    gemm128_wmma<true, true, true><<<gGemm, TPB, 0, stream>>>(
        act16, Bp + (size_t)3 * WPACK, mlp_b, nullptr, h16b);
    gemm128_wmma<true, true, false><<<gGemm, TPB, 0, stream>>>(
        h16b, Bp + (size_t)4 * WPACK, mlp_b + DIM, agg, nullptr);
    head_k<<<gNode, TPB, 0, stream>>>(agg, out_w, out_b, out, NNODES);
    (void)in_sizes; (void)n_in; (void)out_size; (void)ws_size;
}
